// RPN_26431228739782
// MI455X (gfx1250) — matrix operand order repaired and checked by reading, not compiled, not run on hardware
//
#include <hip/hip_runtime.h>
#include <hip/hip_bf16.h>
#include <math.h>

typedef _Float16 v16h __attribute__((ext_vector_type(16)));
typedef _Float16 v8h  __attribute__((ext_vector_type(8)));
typedef float    v8f  __attribute__((ext_vector_type(8)));
typedef unsigned tdm_v4u __attribute__((ext_vector_type(4)));
typedef int      tdm_v8i __attribute__((ext_vector_type(8)));
typedef int      tdm_v4i __attribute__((ext_vector_type(4)));

#define NLVL 5
#define Cc 256
#define Bb 2
#define IMGF 800.0f
#define NEGF (-1e9f)
#define NMS_T 0.7f
#define SCALE_CLAMP_F 4.135166556742356f   /* log(1000/16) */
#define POST_K 1000
#define TOT_P 53294
#define TOTS 159882     /* 3*TOT_P  */
#define TOTD 639528     /* 12*TOT_P */
#define NCAND 4507      /* 1000*4 + 507 */

// compile-time per-level tables (usable inside templated device code)
static constexpr int SOFF_A[5] = {0, 120000, 150000, 157500, 159375};

__constant__ int   c_H[5]      = {200,100,50,25,13};
__constant__ int   c_W[5]      = {200,100,50,25,13};
__constant__ int   c_HW[5]     = {40000,10000,2500,625,169};
__constant__ int   c_POFF[5]   = {0,40000,50000,52500,53125};
__constant__ int   c_SOFF[5]   = {0,120000,150000,157500,159375};
__constant__ float c_STRIDE[5] = {4.f,8.f,16.f,32.f,64.f};
__constant__ float c_SIZE[5]   = {32.f,64.f,128.f,256.f,512.f};
__constant__ int   c_K[5]      = {1000,1000,1000,1000,507};
__constant__ int   c_COFF[5]   = {0,1000,2000,3000,4000};

__device__ __forceinline__ unsigned fmap(float s) {
  unsigned u = __float_as_uint(s);
  return (u & 0x80000000u) ? ~u : (u | 0x80000000u);
}
__device__ __forceinline__ float funmap(unsigned u) {
  return (u & 0x80000000u) ? __uint_as_float(u & 0x7FFFFFFFu) : __uint_as_float(~u);
}

// ---------------------------------------------------------------- weights f32 -> f16 [kyx][co][ci]
__global__ void k_wcvt(const float* __restrict__ w, _Float16* __restrict__ w16) {
  int i = blockIdx.x * 256 + threadIdx.x;            // over 9*256*256
  if (i >= 9 * 256 * 256) return;
  int ci = i & 255, co = (i >> 8) & 255, kyx = i >> 16;
  w16[(kyx * 256 + co) * 256 + ci] = (_Float16)w[(co * 256 + ci) * 9 + kyx];
}

// ---------------------------------------------------------------- seed head buffers with biases
__global__ void k_init(float* __restrict__ scores, float* __restrict__ deltas,
                       const float* __restrict__ obj_b, const float* __restrict__ delta_b) {
  int i = blockIdx.x * 256 + threadIdx.x;
  if (i >= Bb * TOT_P) return;
  int b = i / TOT_P, pa = i % TOT_P;
  int lvl = 4;
  for (int l = 0; l < 4; ++l) if (pa < c_POFF[l + 1]) { lvl = l; break; }
  int p = pa - c_POFF[lvl];
  float* s = scores + (size_t)b * TOTS + c_SOFF[lvl] + (size_t)p * 3;
  for (int a = 0; a < 3; ++a) s[a] = obj_b[a];
  float* d = deltas + (size_t)b * TOTD + 4 * (size_t)c_SOFF[lvl] + (size_t)p * 12;
  for (int r = 0; r < 12; ++r) d[r] = delta_b[r];
}

// ---------------------------------------------------------------- fused conv3x3+relu + 1x1 heads
// One instantiation per FPN level (H,W compile-time -> feature channel strides become
// 24-bit instruction offsets).  WG tile: 64 co x 128 px; wave = 16co x 64px ->
// 4 x v_wmma_f32_16x16x32_f16 per 32-wide K chunk.  Weight tile staged by the
// Tensor Data Mover when available (tensor_load_to_lds + s_wait_tensorcnt).
template <int LVL, int H, int W>
__launch_bounds__(256)
__global__ void k_conv(const float* __restrict__ feat,
                       const _Float16* __restrict__ w16,
                       const float* __restrict__ conv_b,
                       const float* __restrict__ obj_w,
                       const float* __restrict__ delta_w,
                       float* __restrict__ scores, float* __restrict__ deltas) {
  constexpr int HW   = H * W;
  constexpr int PXTL = (HW + 127) / 128;

  __shared__ __align__(32) _Float16 sA[64][32];    // [co][ci]
  __shared__ __align__(32) _Float16 sB[128][32];   // [px][ci]
  __shared__ float tOut[64][129];                  // [co][px] relu'd conv out

  int bx = blockIdx.x;
  int b   = bx / (PXTL * 4);
  int r   = bx % (PXTL * 4);
  int cot = r & 3;
  int p0  = (r >> 2) * 128;
  int co_base = cot * 64;

  int tid  = threadIdx.x;
  int lane = tid & 31, wid = tid >> 5;
  int co_blk = wid & 3, px_blk = wid >> 2;   // 4 co blocks x 2 px blocks
  int m = lane & 15, half = lane >> 4;

  // feature staging role: 128 rows x 2 groups of 16
  int srF = tid >> 1, sgF = (tid & 1) * 16;
  // weight staging role (fallback path): 64 rows x 4 groups of 8
  int srW = tid >> 2, sgW = (tid & 3) * 8;
  (void)srW; (void)sgW;

  int pS = p0 + srF;
  bool pv = pS < HW;
  int yS = pv ? pS / W : 0;
  int xS = pv ? pS % W : 0;

  v8f acc[4] = {v8f{}, v8f{}, v8f{}, v8f{}};

  for (int kyx = 0; kyx < 9; ++kyx) {
    int dy = kyx / 3 - 1, dx = kyx % 3 - 1;
    int yy = yS + dy, xx = xS + dx;
    bool inb = pv && yy >= 0 && yy < H && xx >= 0 && xx < W;
    int q = yy * W + xx;
    int qc = q < 0 ? 0 : (q >= HW ? HW - 1 : q);   // clamp: loads always legal
    const float* fpt = feat + (size_t)b * Cc * HW + (size_t)sgF * HW + qc;
    const _Float16* wrow = w16 + ((size_t)(kyx * Cc + co_base + srW)) * Cc + sgW;
    (void)wrow;
    #pragma unroll 2
    for (int cc = 0; cc < 8; ++cc) {
      int ci0 = cc * 32;
      // ---- stage A (64co x 32ci f16 tile, row stride 256) ----
#if defined(__HIP_DEVICE_COMPILE__) && __has_builtin(__builtin_amdgcn_tensor_load_to_lds)
      if (wid == 0) {
        unsigned long long ga =
            (unsigned long long)(const void*)(w16 + (size_t)(kyx * Cc + co_base) * Cc + ci0);
        unsigned ldsa = (unsigned)(unsigned long long)(const void*)&sA[0][0];
        tdm_v4u g0 = { 1u,                                    // count=1, user mode
                       ldsa,                                  // lds_addr
                       (unsigned)ga,                          // global_addr[31:0]
                       (unsigned)((ga >> 32) & 0x01FFFFFFu) | 0x80000000u }; // addr[56:32] | type=2
        tdm_v8i g1 = { 0x00010000,        // data_size=1 (2 bytes), no mask/pad/iterate
                       256 << 16,         // tensor_dim0 = 256
                       256 << 16,         // tensor_dim1 = 256
                       32  << 16,         // tile_dim0   = 32
                       64,                // tile_dim1   = 64 (tile_dim2 = 0)
                       256,               // tensor_dim0_stride = 256
                       0,                 // dim1_stride low bits
                       1 };               // dim1_stride = 65536
        tdm_v4i g2 = { 0, 0, 0, 0 };
        tdm_v4i g3 = { 0, 0, 0, 0 };
#if __clang_major__ >= 23
        tdm_v8i g4 = { 0, 0, 0, 0, 0, 0, 0, 0 };
        __builtin_amdgcn_tensor_load_to_lds(g0, g1, g2, g3, g4, 0);
#else
        __builtin_amdgcn_tensor_load_to_lds(g0, g1, g2, g3, 0);
#endif
      }
#else
      *(v8h*)&sA[srW][sgW] = *(const v8h*)&wrow[ci0];
#endif
      // ---- stage B (features): 16 unconditional clamped loads, masked select ----
      _Float16 vloc[16];
      const float* fpc = fpt + (size_t)ci0 * HW;
      #pragma unroll
      for (int e = 0; e < 16; ++e) {
        float v = fpc[(size_t)e * HW];     // offset e*HW*4 bytes: compile-time immediate
        vloc[e] = (_Float16)(inb ? v : 0.f);
      }
      *(v8h*)&sB[srF][sgF]     = *(v8h*)&vloc[0];
      *(v8h*)&sB[srF][sgF + 8] = *(v8h*)&vloc[8];
#if defined(__HIP_DEVICE_COMPILE__) && __has_builtin(__builtin_amdgcn_tensor_load_to_lds)
      if (wid == 0) __builtin_amdgcn_s_wait_tensorcnt(0);
#endif
      __syncthreads();
      // A fragment (16x32 f16 layout): lanes 0-15 K=0..15, lanes 16-31 K=16..31
      v8h alo = *(const v8h*)&sA[co_blk * 16 + m][half * 16];
      v8h ahi = *(const v8h*)&sA[co_blk * 16 + m][half * 16 + 8];
      v16h afr = __builtin_shufflevector(alo, ahi, 0,1,2,3,4,5,6,7,8,9,10,11,12,13,14,15);
      // B fragments (32x16): lanes 0-15 K=0..15, lanes 16-31 K=16..31
      #pragma unroll
      for (int s = 0; s < 4; ++s) {
        v16h bf = *(const v16h*)&sB[px_blk * 64 + s * 16 + m][half * 16];
        acc[s] = __builtin_amdgcn_wmma_f32_16x16x32_f16(false, afr, false, bf,
                                                        (short)0, acc[s], false, false);
      }
      __syncthreads();
    }
  }

  // epilogue: bias + relu into LDS (C/D layout: vgpr rr -> M = rr + 8*half, lane%16 -> N)
  #pragma unroll
  for (int rr = 0; rr < 8; ++rr) {
    int col = co_blk * 16 + half * 8 + rr;
    float bias = conv_b[co_base + col];
    #pragma unroll
    for (int s = 0; s < 4; ++s)
      tOut[col][px_blk * 64 + s * 16 + m] = fmaxf(acc[s][rr] + bias, 0.f);
  }
  __syncthreads();

  // fused 1x1 heads: partial 64-channel dot products, f32 atomics (no t round-trip to HBM)
  if (tid < 128) {
    int p = p0 + tid;
    if (p < HW) {
      float so[3]  = {0.f, 0.f, 0.f};
      float dl[12] = {0.f,0.f,0.f,0.f,0.f,0.f,0.f,0.f,0.f,0.f,0.f,0.f};
      for (int co = 0; co < 64; ++co) {
        float tv = tOut[co][tid];
        int cog = co_base + co;
        #pragma unroll
        for (int a = 0; a < 3; ++a)  so[a] += tv * obj_w[a * Cc + cog];
        #pragma unroll
        for (int q2 = 0; q2 < 12; ++q2) dl[q2] += tv * delta_w[q2 * Cc + cog];
      }
      float* sp = scores + (size_t)b * TOTS + SOFF_A[LVL] + (size_t)p * 3;
      #pragma unroll
      for (int a = 0; a < 3; ++a) atomicAdd(&sp[a], so[a]);
      float* dp = deltas + (size_t)b * TOTD + 4 * (size_t)SOFF_A[LVL] + (size_t)p * 12;
      #pragma unroll
      for (int q2 = 0; q2 < 12; ++q2) atomicAdd(&dp[q2], dl[q2]);
    }
  }
}

// ---------------------------------------------------------------- bitonic sort (desc key, asc val ties)
__device__ void bitonic_desc(unsigned* key, int* val, int N, int tid, int nth) {
  for (int k = 2; k <= N; k <<= 1) {
    for (int j = k >> 1; j > 0; j >>= 1) {
      for (int i = tid; i < N; i += nth) {
        int ixj = i ^ j;
        if (ixj > i) {
          unsigned ka = key[i], kb = key[ixj];
          int va = val[i], vb = val[ixj];
          bool aFirst = (ka > kb) || (ka == kb && va < vb);
          bool up = ((i & k) == 0);
          if (up ? !aFirst : aFirst) { key[i] = kb; key[ixj] = ka; val[i] = vb; val[ixj] = va; }
        }
      }
      __syncthreads();
    }
  }
}

// ---------------------------------------------------------------- per-(b,level) top-k + box decode
__launch_bounds__(1024)
__global__ void k_topk(const float* __restrict__ scores, const float* __restrict__ deltas,
                       float* __restrict__ cand_boxes, float* __restrict__ cand_scores) {
  __shared__ int hist[256];
  __shared__ unsigned skey[2048];
  __shared__ int sval[2048];
  __shared__ int s_b1, s_base, s_cnt;
  __shared__ unsigned s_T;

  int b = blockIdx.x / NLVL, lvl = blockIdx.x % NLVL;
  int M = c_HW[lvl] * 3, K = c_K[lvl], W = c_W[lvl];
  const float* sc = scores + (size_t)b * TOTS + c_SOFF[lvl];
  int tid = threadIdx.x, nth = blockDim.x;

  if (tid < 256) hist[tid] = 0;
  __syncthreads();
  for (int i = tid; i < M; i += nth) atomicAdd(&hist[fmap(sc[i]) >> 24], 1);
  __syncthreads();
  if (tid == 0) {
    int cum = 0; s_b1 = 0; s_base = 0;
    for (int bb = 255; bb >= 0; --bb) { cum += hist[bb]; if (cum >= K) { s_b1 = bb; s_base = cum - hist[bb]; break; } }
  }
  __syncthreads();
  int b1 = s_b1, base = s_base;
  if (tid < 256) hist[tid] = 0;
  __syncthreads();
  for (int i = tid; i < M; i += nth) {
    unsigned u = fmap(sc[i]);
    if ((int)(u >> 24) == b1) atomicAdd(&hist[(u >> 16) & 255], 1);
  }
  __syncthreads();
  if (tid == 0) {
    int cum = base; unsigned T = (unsigned)b1 << 24;
    for (int bb = 255; bb >= 0; --bb) { cum += hist[bb]; if (cum >= K) { T = ((unsigned)b1 << 24) | ((unsigned)bb << 16); break; } }
    s_T = T; s_cnt = 0;
  }
  __syncthreads();
  unsigned T = s_T;
  for (int i = tid; i < M; i += nth) {
    unsigned u = fmap(sc[i]);
    if (u >= T) { int pos = atomicAdd(&s_cnt, 1); if (pos < 2048) { skey[pos] = u; sval[pos] = i; } }
  }
  __syncthreads();
  int cnt = s_cnt; if (cnt > 2048) cnt = 2048;
  for (int i = tid; i < 2048; i += nth) if (i >= cnt) { skey[i] = 0u; sval[i] = 0x7FFFFFFF; }
  __syncthreads();
  bitonic_desc(skey, sval, 2048, tid, nth);

  float stride = c_STRIDE[lvl], size = c_SIZE[lvl];
  const float* dbase = deltas + (size_t)b * TOTD + 4 * (size_t)c_SOFF[lvl];
  for (int j = tid; j < K; j += nth) {
    int idx = sval[j];
    int p = idx / 3, a = idx % 3;
    int y = p / W, x = p % W;
    float arv = (a == 0) ? 0.5f : ((a == 1) ? 1.f : 2.f);
    float sq = sqrtf(arv);
    float aw = size / sq, ah = size * sq;
    const float* dl = dbase + (size_t)p * 12 + a * 4;
    float ddx = dl[0], ddy = dl[1];
    float ddw = fminf(dl[2], SCALE_CLAMP_F), ddh = fminf(dl[3], SCALE_CLAMP_F);
    float cx = (float)x * stride, cy = (float)y * stride;
    float pcx = ddx * aw + cx, pcy = ddy * ah + cy;
    float pw = expf(ddw) * aw, ph = expf(ddh) * ah;
    int o = (b * NCAND + c_COFF[lvl] + j) * 4;
    cand_boxes[o + 0] = pcx - 0.5f * pw;
    cand_boxes[o + 1] = pcy - 0.5f * ph;
    cand_boxes[o + 2] = pcx + 0.5f * pw;
    cand_boxes[o + 3] = pcy + 0.5f * ph;
    cand_scores[b * NCAND + c_COFF[lvl] + j] = funmap(skey[j]);
  }
}

// ---------------------------------------------------------------- per-image sort of 4507 candidates
__launch_bounds__(1024)
__global__ void k_sort(const float* __restrict__ cand_boxes, const float* __restrict__ cand_scores,
                       float* __restrict__ s_boxes, float* __restrict__ s_scores, int* __restrict__ s_lvl) {
  extern __shared__ unsigned char dynsm[];
  unsigned* key = (unsigned*)dynsm;
  int* val = (int*)(dynsm + 8192 * sizeof(unsigned));
  int b = blockIdx.x, tid = threadIdx.x, nth = blockDim.x;
  for (int i = tid; i < 8192; i += nth) {
    if (i < NCAND) {
      const float* bx = cand_boxes + (size_t)(b * NCAND + i) * 4;
      float x1 = fminf(fmaxf(bx[0], 0.f), IMGF), y1 = fminf(fmaxf(bx[1], 0.f), IMGF);
      float x2 = fminf(fmaxf(bx[2], 0.f), IMGF), y2 = fminf(fmaxf(bx[3], 0.f), IMGF);
      bool valid = (x2 - x1 > 0.f) && (y2 - y1 > 0.f);
      float s = valid ? cand_scores[b * NCAND + i] : NEGF;
      key[i] = fmap(s); val[i] = i;
    } else { key[i] = 0u; val[i] = 0x7FFFFFFF; }
  }
  __syncthreads();
  bitonic_desc(key, val, 8192, tid, nth);
  for (int i = tid; i < NCAND; i += nth) {
    int v = val[i];
    const float* bx = cand_boxes + (size_t)(b * NCAND + v) * 4;
    float* ob = s_boxes + (size_t)(b * NCAND + i) * 4;
    ob[0] = fminf(fmaxf(bx[0], 0.f), IMGF);
    ob[1] = fminf(fmaxf(bx[1], 0.f), IMGF);
    ob[2] = fminf(fmaxf(bx[2], 0.f), IMGF);
    ob[3] = fminf(fmaxf(bx[3], 0.f), IMGF);
    s_scores[b * NCAND + i] = funmap(key[i]);
    s_lvl[b * NCAND + i] = v / 1000;
  }
}

// ---------------------------------------------------------------- per-image greedy NMS + output
__launch_bounds__(1024)
__global__ void k_nms(const float* __restrict__ s_boxes, const float* __restrict__ s_scores,
                      const int* __restrict__ s_lvl, float* __restrict__ out) {
  extern __shared__ unsigned char dynsm[];
  float* X1 = (float*)dynsm;
  float* Y1 = X1 + NCAND; float* X2 = Y1 + NCAND; float* Y2 = X2 + NCAND;
  int* keep = (int*)(Y2 + NCAND);
  int b = blockIdx.x, tid = threadIdx.x, nth = blockDim.x;

  for (int i = tid; i < NCAND; i += nth) {
    const float* bx = s_boxes + (size_t)(b * NCAND + i) * 4;
    float off = (float)s_lvl[b * NCAND + i] * (IMGF + 1.f);
    X1[i] = bx[0] + off; Y1[i] = bx[1] + off;
    X2[i] = bx[2] + off; Y2[i] = bx[3] + off;
    keep[i] = (s_scores[b * NCAND + i] > NEGF * 0.5f) ? 1 : 0;
  }
  __syncthreads();

  for (int i = 0; i < NCAND - 1; ++i) {
    if (keep[i]) {
      float x1 = X1[i], y1 = Y1[i], x2 = X2[i], y2 = Y2[i];
      float area_i = (x2 - x1) * (y2 - y1);
      for (int j = i + 1 + tid; j < NCAND; j += nth) {
        if (keep[j]) {
          float ix1 = fmaxf(x1, X1[j]), iy1 = fmaxf(y1, Y1[j]);
          float ix2 = fminf(x2, X2[j]), iy2 = fminf(y2, Y2[j]);
          float iw = fmaxf(ix2 - ix1, 0.f), ih = fmaxf(iy2 - iy1, 0.f);
          float inter = iw * ih;
          float area_j = (X2[j] - X1[j]) * (Y2[j] - Y1[j]);
          float iou = inter / fmaxf(area_i + area_j - inter, 1e-9f);
          if (iou > NMS_T) keep[j] = 0;
        }
      }
    }
    __syncthreads();
  }

  if (tid == 0) {
    float* op = out + (size_t)b * POST_K * 5;
    int cnt = 0;
    for (int i = 0; i < NCAND && cnt < POST_K; ++i) if (keep[i]) {
      const float* bx = s_boxes + (size_t)(b * NCAND + i) * 4;
      op[cnt * 5 + 0] = bx[0]; op[cnt * 5 + 1] = bx[1];
      op[cnt * 5 + 2] = bx[2]; op[cnt * 5 + 3] = bx[3];
      op[cnt * 5 + 4] = s_scores[b * NCAND + i];
      ++cnt;
    }
    for (int i = 0; i < NCAND && cnt < POST_K; ++i) if (!keep[i]) {
      const float* bx = s_boxes + (size_t)(b * NCAND + i) * 4;
      op[cnt * 5 + 0] = bx[0]; op[cnt * 5 + 1] = bx[1];
      op[cnt * 5 + 2] = bx[2]; op[cnt * 5 + 3] = bx[3];
      op[cnt * 5 + 4] = NEGF;
      ++cnt;
    }
  }
}

// ---------------------------------------------------------------- workspace layout
static constexpr size_t align256(size_t x) { return (x + 255) & ~(size_t)255; }
static constexpr size_t OFF_W16 = 0;
static constexpr size_t SZ_W16  = (size_t)9 * 256 * 256 * 2;
static constexpr size_t OFF_SC  = align256(OFF_W16 + SZ_W16);
static constexpr size_t SZ_SC   = (size_t)Bb * TOTS * 4;
static constexpr size_t OFF_DL  = align256(OFF_SC + SZ_SC);
static constexpr size_t SZ_DL   = (size_t)Bb * TOTD * 4;
static constexpr size_t OFF_CB  = align256(OFF_DL + SZ_DL);
static constexpr size_t SZ_CB   = (size_t)Bb * NCAND * 4 * 4;
static constexpr size_t OFF_CS  = align256(OFF_CB + SZ_CB);
static constexpr size_t SZ_CS   = (size_t)Bb * NCAND * 4;
static constexpr size_t OFF_SB  = align256(OFF_CS + SZ_CS);
static constexpr size_t OFF_SS  = align256(OFF_SB + SZ_CB);
static constexpr size_t OFF_SL  = align256(OFF_SS + SZ_CS);

template <int LVL, int H, int W>
static void launch_conv(const float* f, const _Float16* w16, const float* conv_b,
                        const float* obj_w, const float* delta_w,
                        float* scores, float* deltas, hipStream_t stream) {
  constexpr int PXTL = (H * W + 127) / 128;
  k_conv<LVL, H, W><<<PXTL * 4 * Bb, 256, 0, stream>>>(f, w16, conv_b, obj_w, delta_w,
                                                       scores, deltas);
}

extern "C" void kernel_launch(void* const* d_in, const int* in_sizes, int n_in,
                              void* d_out, int out_size, void* d_ws, size_t ws_size,
                              hipStream_t stream) {
  (void)in_sizes; (void)n_in; (void)out_size; (void)ws_size;
  const float* f0      = (const float*)d_in[0];
  const float* f1      = (const float*)d_in[1];
  const float* f2      = (const float*)d_in[2];
  const float* f3      = (const float*)d_in[3];
  const float* f4      = (const float*)d_in[4];
  const float* conv_w  = (const float*)d_in[5];
  const float* conv_b  = (const float*)d_in[6];
  const float* obj_w   = (const float*)d_in[7];
  const float* obj_b   = (const float*)d_in[8];
  const float* delta_w = (const float*)d_in[9];
  const float* delta_b = (const float*)d_in[10];

  char* ws = (char*)d_ws;
  _Float16* w16      = (_Float16*)(ws + OFF_W16);
  float* scores      = (float*)(ws + OFF_SC);
  float* deltas      = (float*)(ws + OFF_DL);
  float* cand_boxes  = (float*)(ws + OFF_CB);
  float* cand_scores = (float*)(ws + OFF_CS);
  float* s_boxes     = (float*)(ws + OFF_SB);
  float* s_scores    = (float*)(ws + OFF_SS);
  int*   s_lvl       = (int*)(ws + OFF_SL);
  float* out         = (float*)d_out;

  k_wcvt<<<(9 * 256 * 256 + 255) / 256, 256, 0, stream>>>(conv_w, w16);
  k_init<<<(Bb * TOT_P + 255) / 256, 256, 0, stream>>>(scores, deltas, obj_b, delta_b);
  launch_conv<0, 200, 200>(f0, w16, conv_b, obj_w, delta_w, scores, deltas, stream);
  launch_conv<1, 100, 100>(f1, w16, conv_b, obj_w, delta_w, scores, deltas, stream);
  launch_conv<2,  50,  50>(f2, w16, conv_b, obj_w, delta_w, scores, deltas, stream);
  launch_conv<3,  25,  25>(f3, w16, conv_b, obj_w, delta_w, scores, deltas, stream);
  launch_conv<4,  13,  13>(f4, w16, conv_b, obj_w, delta_w, scores, deltas, stream);
  k_topk<<<Bb * NLVL, 1024, 0, stream>>>(scores, deltas, cand_boxes, cand_scores);
  k_sort<<<Bb, 1024, 8192 * 8, stream>>>(cand_boxes, cand_scores, s_boxes, s_scores, s_lvl);
  k_nms<<<Bb, 1024, (size_t)NCAND * 4 * 5, stream>>>(s_boxes, s_scores, s_lvl, out);
}
